// AP_64338610094239
// MI455X (gfx1250) — compile-verified, matrix operand order checked by reading, not run on hardware
//
#include <hip/hip_runtime.h>
#include <hip/hip_bf16.h>

typedef __attribute__((ext_vector_type(16))) _Float16 v16h;
typedef __attribute__((ext_vector_type(8)))  float    v8f;

#define NTH 10
#define NCNT 21                          // rows: 0..9 tp[t], 10..19 cnt[t], 20 P
#define S_PER_BATCH (96 * 256 * 256)     // 6,291,456 voxels per batch
#define N_TOTAL     (2 * S_PER_BATCH)    // 12,582,912 voxels
#define BLOCKS 1024
#define TPB 256

__global__ void ap_init(float* ws) {
    int t = threadIdx.x;
    if (t < NCNT) ws[t] = 0.0f;
}

__global__ __launch_bounds__(TPB) void ap_main(const float* __restrict__ y_pred,
                                               const int*   __restrict__ y_true,
                                               const float* __restrict__ thr,
                                               float*       __restrict__ ws) {
    // thresholds are uniform -> scalar loads
    float th[NTH];
    #pragma unroll
    for (int t = 0; t < NTH; ++t) th[t] = thr[t];

    // acc[t]: low 16 bits = cnt[t] (s > th[t]), high 16 bits = tp[t] (s > th[t] && y==1)
    unsigned acc[NTH];
    #pragma unroll
    for (int t = 0; t < NTH; ++t) acc[t] = 0u;
    unsigned cPos = 0u;

    const int nChunks = N_TOTAL / 4;       // float4 / int4 chunks
    const int sChunks = S_PER_BATCH / 4;
    const int stride  = gridDim.x * blockDim.x;

    for (int i = blockIdx.x * blockDim.x + threadIdx.x; i < nChunks; i += stride) {
        // class-1 plane of batch b starts at element (2b+1)*S -> float4 index (2b+1)*sChunks
        int b = (i >= sChunks) ? 1 : 0;
        float4 sv = ((const float4*)y_pred)[i + (b + 1) * sChunks];
        int4   yv = ((const int4*)y_true)[i];

        int inext = i + stride;
        if (inext < nChunks) {
            int bn = (inext >= sChunks) ? 1 : 0;
            __builtin_prefetch((const void*)(((const float4*)y_pred) + inext + (bn + 1) * sChunks), 0, 3);
            __builtin_prefetch((const void*)(((const int4*)y_true) + inext), 0, 3);
        }

        float ss[4] = {sv.x, sv.y, sv.z, sv.w};
        int   yy[4] = {yv.x, yv.y, yv.z, yv.w};
        #pragma unroll
        for (int v = 0; v < 4; ++v) {
            unsigned isP = (yy[v] == 1) ? 1u : 0u;
            cPos += isP;
            unsigned add = 1u + (isP << 16);
            float s = ss[v];
            #pragma unroll
            for (int t = 0; t < NTH; ++t)
                acc[t] += (s > th[t]) ? add : 0u;
        }
    }

    // Stage per-thread counters to LDS as f16 (counts <= 48 -> exact)
    __shared__ _Float16 lds[NCNT][TPB];
    int tid = threadIdx.x;
    #pragma unroll
    for (int t = 0; t < NTH; ++t) {
        lds[t][tid]       = (_Float16)(float)(acc[t] >> 16);      // tp[t]   -> row t
        lds[NTH + t][tid] = (_Float16)(float)(acc[t] & 0xFFFFu);  // cnt[t]  -> row 10+t
    }
    lds[20][tid] = (_Float16)(float)cPos;                         // P       -> row 20
    __syncthreads();

    // Wave 0 reduces all 256 threads' counters with WMMA(A, ones):
    // D[m,*] = sum_k A[m,k]; lane L feeds row L%16, so rows = counter index.
    if (tid < 32) {
        int lane = tid;
        int m = lane & 15, h = lane >> 4;

        v16h ones;
        #pragma unroll
        for (int e = 0; e < 16; ++e) ones[e] = (_Float16)1.0f;

        v8f accA = {};  // rows 0..15  -> counters 0..15
        v8f accB = {};  // rows 0..4   -> counters 16..20
        #pragma unroll
        for (int j = 0; j < 8; ++j) {
            int col = 32 * j + 16 * h;   // lanes m and m+16 together cover 32 threads/step
            v16h a1, a2;
            #pragma unroll
            for (int e = 0; e < 16; ++e) a1[e] = lds[m][col + e];
            #pragma unroll
            for (int e = 0; e < 16; ++e) a2[e] = (_Float16)0.0f;
            if (m < 5) {
                #pragma unroll
                for (int e = 0; e < 16; ++e) a2[e] = lds[16 + m][col + e];
            }
            accA = __builtin_amdgcn_wmma_f32_16x16x32_f16(false, a1, false, ones,
                                                          (short)0, accA, false, false);
            accB = __builtin_amdgcn_wmma_f32_16x16x32_f16(false, a2, false, ones,
                                                          (short)0, accB, false, false);
        }

        // C/D layout: VGPR r holds row r (lanes 0-15) and row 8+r (lanes 16-31);
        // every column is identical (B = ones), so read column 0 / 16.
        if (lane == 0) {
            #pragma unroll
            for (int r = 0; r < 8; ++r) atomicAdd(&ws[r], accA[r]);        // counters 0..7
            #pragma unroll
            for (int r = 0; r < 5; ++r) atomicAdd(&ws[16 + r], accB[r]);   // counters 16..20
        } else if (lane == 16) {
            #pragma unroll
            for (int r = 0; r < 8; ++r) atomicAdd(&ws[8 + r], accA[r]);    // counters 8..15
        }
    }
}

__global__ void ap_final(const float* __restrict__ ws, float* __restrict__ out) {
    if (threadIdx.x != 0 || blockIdx.x != 0) return;
    const float EPS = 1e-7f;
    float P = ws[20];
    float p[NTH + 2], r[NTH + 2];
    p[0] = 0.0f; r[0] = 1.0f;
    for (int t = 0; t < NTH; ++t) {
        float tp  = ws[t];
        float cnt = ws[10 + t];
        float fp  = cnt - tp;                 // cnt = tp + fp
        p[t + 1] = (tp + EPS) / (tp + fp + EPS);
        r[t + 1] = (tp + EPS) / (P + EPS);    // tp + fn == P
    }
    p[NTH + 1] = 1.0f; r[NTH + 1] = 0.0f;
    float area = 0.0f;
    for (int i = 0; i < NTH + 1; ++i)
        area += (r[i + 1] - r[i]) * (p[i + 1] + p[i]);
    area *= 0.5f;
    out[0] = fabsf(area);
}

extern "C" void kernel_launch(void* const* d_in, const int* in_sizes, int n_in,
                              void* d_out, int out_size, void* d_ws, size_t ws_size,
                              hipStream_t stream) {
    const float* y_pred = (const float*)d_in[0];
    const int*   y_true = (const int*)d_in[1];
    const float* thr    = (const float*)d_in[2];
    float* ws = (float*)d_ws;   // 21 float counters

    ap_init<<<1, 32, 0, stream>>>(ws);
    ap_main<<<BLOCKS, TPB, 0, stream>>>(y_pred, y_true, thr, ws);
    ap_final<<<1, 1, 0, stream>>>(ws, (float*)d_out);
}
